// PointNetSetAbstraction_66176856096773
// MI455X (gfx1250) — matrix operand order repaired and checked by reading, not compiled, not run on hardware
//
#include <hip/hip_runtime.h>
#include <hip/hip_bf16.h>

typedef __attribute__((ext_vector_type(16))) _Float16 v16h;
typedef __attribute__((ext_vector_type(8)))  _Float16 v8h;
typedef __attribute__((ext_vector_type(8)))  float    v8f;

#define BATCH   16
#define NPTS    8192
#define NCENT   512
#define KNN     32
#define ROWS    (BATCH * NCENT * KNN)      // 262144
#define R2      0.04f
#define NEWXYZ_ELEMS (BATCH * NCENT * 3)   // 24576

// ---------------- 0: zero BN accumulators ----------------
__global__ void zero_acc_kernel(float* acc) {
    int t = threadIdx.x;
    if (t < 768) acc[t] = 0.0f;
}

// ---------------- 1: farthest point sampling (1 block / batch) ----------------
__global__ __launch_bounds__(1024)
void fps_kernel(const float* __restrict__ xyz, int* __restrict__ fps_idx,
                float* __restrict__ newxyz) {
    __shared__ float dist[NPTS];
    __shared__ float rv[1024];
    __shared__ int   ri[1024];
    const int b   = blockIdx.x;
    const int tid = threadIdx.x;
    const float* bx = xyz + (size_t)b * NPTS * 3;

    // cache this thread's 8 points in registers
    float px[8], py[8], pz[8];
    for (int j = 0; j < 8; ++j) {
        int p = tid + j * 1024;
        px[j] = bx[p * 3 + 0]; py[j] = bx[p * 3 + 1]; pz[j] = bx[p * 3 + 2];
        dist[p] = 1.0e10f;
    }
    __syncthreads();

    int far = 0;
    for (int it = 0; it < NCENT; ++it) {
        if (tid == 0) {
            fps_idx[b * NCENT + it] = far;
            newxyz[(b * NCENT + it) * 3 + 0] = bx[far * 3 + 0];
            newxyz[(b * NCENT + it) * 3 + 1] = bx[far * 3 + 1];
            newxyz[(b * NCENT + it) * 3 + 2] = bx[far * 3 + 2];
        }
        const float cx = bx[far * 3 + 0], cy = bx[far * 3 + 1], cz = bx[far * 3 + 2];
        float bv = -1.0f; int bi = 0;
        #pragma unroll
        for (int j = 0; j < 8; ++j) {
            int p = tid + j * 1024;
            float dx = px[j] - cx, dy = py[j] - cy, dz = pz[j] - cz;
            float d  = dx * dx + dy * dy + dz * dz;
            float nd = fminf(dist[p], d);
            dist[p] = nd;
            if (nd > bv) { bv = nd; bi = p; }
        }
        rv[tid] = bv; ri[tid] = bi;
        __syncthreads();
        for (int off = 512; off > 0; off >>= 1) {
            if (tid < off) {
                float ov = rv[tid + off]; int oi = ri[tid + off];
                if (ov > rv[tid] || (ov == rv[tid] && oi < ri[tid])) { rv[tid] = ov; ri[tid] = oi; }
            }
            __syncthreads();
        }
        far = ri[0];
        __syncthreads();
    }
}

// ---------------- 2: ball query (1 wave / center) ----------------
__global__ __launch_bounds__(256)
void qball_kernel(const float* __restrict__ xyz, const float* __restrict__ newxyz,
                  int* __restrict__ idx) {
    __shared__ int res[8][KNN];
    const int tid  = threadIdx.x;
    const int wid  = tid >> 5, lane = tid & 31;
    const int pair = blockIdx.x * 8 + wid;          // b*512+s
    const int b    = pair >> 9;
    const float cx = newxyz[pair * 3 + 0];
    const float cy = newxyz[pair * 3 + 1];
    const float cz = newxyz[pair * 3 + 2];
    const float* bx = xyz + (size_t)b * NPTS * 3;

    int count = 0;
    for (int base = 0; base < NPTS && count < KNN; base += 32) {
        int i = base + lane;
        float dx = bx[i * 3 + 0] - cx;
        float dy = bx[i * 3 + 1] - cy;
        float dz = bx[i * 3 + 2] - cz;
        bool m = (dx * dx + dy * dy + dz * dz) <= R2;
        unsigned bal = (unsigned)__ballot(m);
        int pos = count + __popc(bal & ((1u << lane) - 1u));
        if (m && pos < KNN) res[wid][pos] = i;
        count += __popc(bal);
    }
    if (count > KNN) count = KNN;
    int first = (count > 0) ? res[wid][0] : 0;
    int val   = (lane < count) ? res[wid][lane] : first;
    idx[pair * KNN + lane] = val;
}

// ---------------- 3: layer0 (Cin=6 -> 64), gather + scalar matmul + BN stats --------
__global__ __launch_bounds__(256)
void layer0_kernel(const float* __restrict__ xyz, const float* __restrict__ pts,
                   const float* __restrict__ newxyz, const int* __restrict__ idx,
                   const float* __restrict__ w0, const float* __restrict__ b0,
                   _Float16* __restrict__ act0, float* __restrict__ acc) {
    __shared__ float w0s[64 * 6];
    __shared__ float b0s[64];
    __shared__ float ssum[64], ssq[64];
    __shared__ float tile[256 * 33];                 // staging for channel reductions
    const int tid = threadIdx.x;
    for (int i = tid; i < 64 * 6; i += 256) w0s[i] = w0[i];
    if (tid < 64) { b0s[tid] = b0[tid]; ssum[tid] = 0.0f; ssq[tid] = 0.0f; }
    __syncthreads();

    const int r  = blockIdx.x * 256 + tid;           // row id in [0, ROWS)
    const int b  = r >> 14;                          // /(512*32)
    const int bs = r >> 5;                           // b*512+s
    const int pi = idx[r];

    float f[6];
    const float* p = xyz + ((size_t)b * NPTS + pi) * 3;
    f[0] = p[0] - newxyz[bs * 3 + 0];
    f[1] = p[1] - newxyz[bs * 3 + 1];
    f[2] = p[2] - newxyz[bs * 3 + 2];
    const float* q = pts + ((size_t)b * NPTS + pi) * 3;
    f[3] = q[0]; f[4] = q[1]; f[5] = q[2];

    float y[64];
    for (int c = 0; c < 64; ++c) {
        float v = b0s[c];
        #pragma unroll
        for (int j = 0; j < 6; ++j) v += w0s[c * 6 + j] * f[j];
        y[c] = v;
    }

    // vectorized f16 store of the row (8 x b128, immediate offsets)
    v8h* outp = (v8h*)(act0 + (size_t)r * 64);
    #pragma unroll
    for (int j = 0; j < 8; ++j) {
        v8h h;
        #pragma unroll
        for (int e = 0; e < 8; ++e) h[e] = (_Float16)y[j * 8 + e];
        outp[j] = h;
    }

    // channel statistics: stage 32 channels at a time, strided reduce
    float* trow = tile + tid * 33;
    for (int pphase = 0; pphase < 2; ++pphase) {
        __syncthreads();
        #pragma unroll
        for (int c2 = 0; c2 < 32; ++c2) trow[c2] = y[pphase * 32 + c2];
        __syncthreads();
        int c = tid & 31, g = tid >> 5;
        const float* rr = tile + (g * 32) * 33 + c;
        float s1 = 0.0f, s2 = 0.0f;
        #pragma unroll
        for (int rl = 0; rl < 32; ++rl) {
            float v = rr[rl * 33];
            s1 += v; s2 += v * v;
        }
        atomicAdd(&ssum[pphase * 32 + c], s1);
        atomicAdd(&ssq[pphase * 32 + c], s2);
    }
    __syncthreads();
    if (tid < 64) { atomicAdd(&acc[tid], ssum[tid]); atomicAdd(&acc[128 + tid], ssq[tid]); }
}

// ---------------- BN param computation (scale/shift per channel) ----------------
__global__ void bnparams_kernel(const float* __restrict__ acc, const float* __restrict__ g,
                                const float* __restrict__ be, float* __restrict__ bnp,
                                int C, float invM) {
    int c = threadIdx.x;
    if (c < C) {
        float mean = acc[c] * invM;
        float var  = acc[128 + c] * invM - mean * mean;
        float sc   = g[c] * rsqrtf(var + 1e-5f);
        bnp[c]     = sc;
        bnp[C + c] = be[c] - mean * sc;
    }
}

// ---------------- WMMA MLP layer: A=relu(bn(prevY)) [f16], Y = A*W^T + b ----------
// Each wave owns RT=4 row tiles (A kept in registers), loops over N tiles, so every
// packed B fragment (2 x ds_load_b128) feeds 4 WMMA pairs. All global accesses use a
// single per-lane base pointer + compile-time immediate offsets.
template <int COUT>
__global__ __launch_bounds__(256)
void mlp_wmma_kernel(const _Float16* __restrict__ actin, const float* __restrict__ bnp_prev,
                     const float* __restrict__ w, const float* __restrict__ bias,
                     _Float16* __restrict__ actout, float* __restrict__ acc) {
    constexpr int CIN = 64;
    constexpr int NT  = COUT / 16;
    constexpr int RT  = 4;
    __shared__ _Float16 bfrag[CIN * COUT];           // fragment-major packed B = W^T
    __shared__ float sc[CIN], sh[CIN];
    __shared__ float ssum[COUT], ssq[COUT];
    const int tid = threadIdx.x;

    // pack B so each lane's fragment is 16 contiguous f16:
    // bfrag[((nt*2+h)*32 + lane)*16 + e] = W^T[k][n],
//   n = nt*16 + (lane&15), k = 32*h + (lane&16 ? 8 : 0) + (e&7) + (e&8 ? 16 : 0)
    for (int i = tid; i < CIN * COUT; i += 256) {
        int e    = i & 15;
        int lane = (i >> 4) & 31;
        int h    = (i >> 9) & 1;
        int nt   = i >> 10;
        int n    = nt * 16 + (lane & 15);
        int k    = 32 * h + ((lane & 16) ? 8 : 0) + (e & 7) + ((e & 8) ? 16 : 0);
        bfrag[i] = (_Float16)w[n * CIN + k];
    }
    if (tid < CIN) { sc[tid] = bnp_prev[tid]; sh[tid] = bnp_prev[CIN + tid]; }
    for (int i = tid; i < COUT; i += 256) { ssum[i] = 0.0f; ssq[i] = 0.0f; }
    __syncthreads();

    const int lane = tid & 31;
    const int wid  = tid >> 5;
    const int rowbase0 = (blockIdx.x * 8 + wid) * (16 * RT);
    const int hl = lane & 15;
    const bool hi = lane >= 16;
    const int c0 = hi ? 1 : 0;                       // first 8-elem chunk index (K = 8*c0)

    // ---- build A fragments for RT row tiles: 4 x global_load_b128 each,
    //      single base pointer + immediate offsets (rt*128 v8h units per tile) ----
    const v8h* abase = (const v8h*)(actin + (size_t)(rowbase0 + hl) * CIN) + c0;
    __builtin_prefetch(actin + (size_t)(rowbase0 + RT * 16 * 8 + hl) * CIN, 0, 1);
    v16h a[RT][2];
    #pragma unroll
    for (int rt = 0; rt < RT; ++rt) {
        v8h r0 = abase[rt * 128 + 0];
        v8h r1 = abase[rt * 128 + 2];
        v8h r2 = abase[rt * 128 + 4];
        v8h r3 = abase[rt * 128 + 6];
        #pragma unroll
        for (int e = 0; e < 8; ++e) {
            int k0 = c0 * 8 + e;                     // klo + e
            float x;
            x = (float)r0[e] * sc[k0]      + sh[k0];      a[rt][0][e]     = (_Float16)(x > 0.0f ? x : 0.0f);
            x = (float)r1[e] * sc[k0 + 16] + sh[k0 + 16]; a[rt][0][e + 8] = (_Float16)(x > 0.0f ? x : 0.0f);
            x = (float)r2[e] * sc[k0 + 32] + sh[k0 + 32]; a[rt][1][e]     = (_Float16)(x > 0.0f ? x : 0.0f);
            x = (float)r3[e] * sc[k0 + 48] + sh[k0 + 48]; a[rt][1][e + 8] = (_Float16)(x > 0.0f ? x : 0.0f);
        }
    }

    // per-lane output base: all store offsets below are compile-time constants
    _Float16* pout = actout + (size_t)(rowbase0 + (hi ? 8 : 0)) * COUT + hl;

    // ---- N tiles ----
    #pragma unroll
    for (int nt = 0; nt < NT; ++nt) {
        const int nn = nt * 16 + hl;
        const v8h* bp = (const v8h*)(bfrag + ((size_t)(nt * 2 + 0) * 32 + lane) * 16);
        const v8h* bq = (const v8h*)(bfrag + ((size_t)(nt * 2 + 1) * 32 + lane) * 16);
        v16h bb0 = __builtin_shufflevector(bp[0], bp[1], 0, 1, 2, 3, 4, 5, 6, 7,
                                           8, 9, 10, 11, 12, 13, 14, 15);
        v16h bb1 = __builtin_shufflevector(bq[0], bq[1], 0, 1, 2, 3, 4, 5, 6, 7,
                                           8, 9, 10, 11, 12, 13, 14, 15);
        const float bb = bias[nn];
        #pragma unroll
        for (int rt = 0; rt < RT; ++rt) {
            v8f c = {};
            c = __builtin_amdgcn_wmma_f32_16x16x32_f16(false, a[rt][0], false, bb0,
                                                       (short)0, c, false, false);
            c = __builtin_amdgcn_wmma_f32_16x16x32_f16(false, a[rt][1], false, bb1,
                                                       (short)0, c, false, false);
            float s1 = 0.0f, s2 = 0.0f;
            #pragma unroll
            for (int j = 0; j < 8; ++j) {
                float y = c[j] + bb;
                // C/D layout: VGPR j -> row j (lanes 0-15) / row j+8 (lanes 16-31)
                pout[(rt * 16 + j) * COUT + nt * 16] = (_Float16)y;
                s1 += y; s2 += y * y;
            }
            s1 += __shfl_xor(s1, 16, 32);            // lanes l and l+16 share nn
            s2 += __shfl_xor(s2, 16, 32);
            if (!hi) { atomicAdd(&ssum[nn], s1); atomicAdd(&ssq[nn], s2); }
        }
    }
    __syncthreads();
    for (int i = tid; i < COUT; i += 256) {
        atomicAdd(&acc[i], ssum[i]);
        atomicAdd(&acc[128 + i], ssq[i]);
    }
}

// ---------------- final: BN + relu + max over K ----------------
__global__ __launch_bounds__(256)
void maxpool_kernel(const _Float16* __restrict__ act2, const float* __restrict__ bnp2,
                    float* __restrict__ out) {
    const int g  = blockIdx.x * 256 + threadIdx.x;    // 0 .. 16*512*128-1
    const int c  = g & 127;
    const int bs = g >> 7;
    const float sc = bnp2[c], sh = bnp2[128 + c];
    const _Float16* base = act2 + (size_t)bs * KNN * 128 + c;
    float m = 0.0f;                                    // relu output is >= 0
    #pragma unroll
    for (int k = 0; k < KNN; ++k) {
        float y = (float)base[k * 128] * sc + sh;      // immediate offsets
        y = (y > 0.0f ? y : 0.0f);
        m = fmaxf(m, y);
    }
    out[NEWXYZ_ELEMS + g] = m;
}

// ---------------- host launcher ----------------
extern "C" void kernel_launch(void* const* d_in, const int* in_sizes, int n_in,
                              void* d_out, int out_size, void* d_ws, size_t ws_size,
                              hipStream_t stream) {
    const float* xyz   = (const float*)d_in[0];
    const float* pts   = (const float*)d_in[1];
    const float* w0    = (const float*)d_in[2];
    const float* b0    = (const float*)d_in[3];
    const float* g0    = (const float*)d_in[4];
    const float* be0   = (const float*)d_in[5];
    const float* w1    = (const float*)d_in[6];
    const float* b1    = (const float*)d_in[7];
    const float* g1    = (const float*)d_in[8];
    const float* be1   = (const float*)d_in[9];
    const float* w2    = (const float*)d_in[10];
    const float* b2    = (const float*)d_in[11];
    const float* g2    = (const float*)d_in[12];
    const float* be2   = (const float*)d_in[13];
    float* out = (float*)d_out;

    char* ws = (char*)d_ws;
    size_t o = 0;
    int*   fps_idx = (int*)(ws + o);      o += (size_t)BATCH * NCENT * 4;          // 32 KB
    int*   idx     = (int*)(ws + o);      o += (size_t)ROWS * 4;                   // 1 MB
    o = (o + 255) & ~(size_t)255;
    float* acc     = (float*)(ws + o);    o += 3 * 256 * 4;                        // 3 layers x (sum,sq)
    float* bnp     = (float*)(ws + o);    o += 3 * 256 * 4;                        // 3 layers x (scale,shift)
    o = (o + 255) & ~(size_t)255;
    _Float16* act0 = (_Float16*)(ws + o); o += (size_t)ROWS * 64 * 2;              // 32 MB
    _Float16* act1 = (_Float16*)(ws + o); o += (size_t)ROWS * 64 * 2;              // 32 MB
    _Float16* act2 = (_Float16*)(ws + o); o += (size_t)ROWS * 128 * 2;             // 64 MB

    float* newxyz = out;                               // first 24576 floats of output
    const float invM = 1.0f / (float)ROWS;

    zero_acc_kernel<<<1, 768, 0, stream>>>(acc);
    fps_kernel<<<BATCH, 1024, 0, stream>>>(xyz, fps_idx, newxyz);
    qball_kernel<<<(BATCH * NCENT) / 8, 256, 0, stream>>>(xyz, newxyz, idx);
    layer0_kernel<<<ROWS / 256, 256, 0, stream>>>(xyz, pts, newxyz, idx, w0, b0,
                                                  act0, acc + 0 * 256);
    bnparams_kernel<<<1, 128, 0, stream>>>(acc + 0 * 256, g0, be0, bnp + 0 * 256, 64, invM);
    mlp_wmma_kernel<64><<<ROWS / (16 * 4) / 8, 256, 0, stream>>>(act0, bnp + 0 * 256, w1, b1,
                                                                 act1, acc + 1 * 256);
    bnparams_kernel<<<1, 128, 0, stream>>>(acc + 1 * 256, g1, be1, bnp + 1 * 256, 64, invM);
    mlp_wmma_kernel<128><<<ROWS / (16 * 4) / 8, 256, 0, stream>>>(act1, bnp + 1 * 256, w2, b2,
                                                                  act2, acc + 2 * 256);
    bnparams_kernel<<<1, 128, 0, stream>>>(acc + 2 * 256, g2, be2, bnp + 2 * 256, 128, invM);
    maxpool_kernel<<<(BATCH * NCENT * 128) / 256, 256, 0, stream>>>(act2, bnp + 2 * 256, out);
    (void)in_sizes; (void)n_in; (void)out_size; (void)ws_size;
}
